// TraceLSTMCell_44435731645052
// MI455X (gfx1250) — matrix operand order repaired and checked by reading, not compiled, not run on hardware
//
#include <hip/hip_runtime.h>

// ---------------------------------------------------------------------------
// TraceLSTMCell for MI455X (gfx1250): bf16 WMMA GEMMs + fused f32 gate math.
// ---------------------------------------------------------------------------

#if defined(__AMDGCN__) && __has_builtin(__builtin_amdgcn_global_load_async_to_lds_b128) && __has_builtin(__builtin_amdgcn_s_wait_asynccnt)
#define USE_ASYNC 1
#else
#define USE_ASYNC 0
#endif

typedef __attribute__((ext_vector_type(16))) __bf16        v16bf;
typedef __attribute__((ext_vector_type(8)))  float         v8f;
typedef __attribute__((ext_vector_type(4)))  unsigned int  u32x4;
typedef int v4i __attribute__((vector_size(16)));   // matches builtin V4i pointee

struct Frag { u32x4 lo, hi; };   // 32 bytes == one v16bf operand

__device__ __forceinline__ v16bf frag_cast(const Frag& f) {
    return __builtin_bit_cast(v16bf, f);
}

__device__ __forceinline__ unsigned short f2bf(float f) {
    unsigned int u = __builtin_bit_cast(unsigned int, f);
    u += 0x7FFFu + ((u >> 16) & 1u);            // round-to-nearest-even
    return (unsigned short)(u >> 16);
}

__device__ __forceinline__ float sigmoid_(float x)  { return 1.0f / (1.0f + __expf(-x)); }
__device__ __forceinline__ float softsign_(float x) { return x / (1.0f + fabsf(x)); }

// ---------------------------------------------------------------------------
// Pack activations -> Act bf16 [4096 x 4608] = [x(512) | h | c | e | r]
// ---------------------------------------------------------------------------
__global__ void convert_act_kernel(const float* __restrict__ x, const float* __restrict__ h,
                                   const float* __restrict__ c, const float* __restrict__ e,
                                   const float* __restrict__ r, unsigned short* __restrict__ act) {
    unsigned int i   = blockIdx.x * 256u + threadIdx.x;   // exact: 4096*4608 threads
    unsigned int row = i / 4608u;
    unsigned int col = i - row * 4608u;
    float v;
    if (col < 512u)        v = x[row * 512u  + col];
    else if (col < 1536u)  v = h[row * 1024u + (col - 512u)];
    else if (col < 2560u)  v = c[row * 1024u + (col - 1536u)];
    else if (col < 3584u)  v = e[row * 1024u + (col - 2560u)];
    else                   v = r[row * 1024u + (col - 3584u)];
    act[i] = f2bf(v);
}

// ---------------------------------------------------------------------------
// Pack a 1024-column slice of an f32 weight matrix (optionally summed with a
// second slice) into a bf16 [rows x 1024] block. dst already offset by caller.
// ---------------------------------------------------------------------------
__global__ void pack_w_kernel(const float* __restrict__ s1, int ld1, int c1,
                              const float* __restrict__ s2, int ld2, int c2,
                              unsigned short* __restrict__ dst) {
    unsigned int i  = blockIdx.x * 256u + threadIdx.x;   // exact: rows*1024 threads
    unsigned int rr = i >> 10;
    unsigned int n  = i & 1023u;
    float v = s1[(size_t)rr * (size_t)ld1 + (size_t)c1 + n];
    if (s2) v += s2[(size_t)rr * (size_t)ld2 + (size_t)c2 + n];
    dst[i] = f2bf(v);
}

// ---------------------------------------------------------------------------
// bf16 WMMA GEMM:  C[4096 x 1024] = A[4096 x K] (ld=lda) * W[K x 1024]
//                                  + bias(biasA+biasB per column) + Cinit
// Block: 128(M) x 64(N), 8 waves of 32, each wave -> 32x32 (2x2 WMMA tiles).
// K-step 32, double-buffered LDS; W transposed into LDS at staging time.
// ---------------------------------------------------------------------------
#define LDSA 40   // padded row stride (ushorts) for 32-wide K tiles

__global__ __launch_bounds__(256) void gemm_bf16_kernel(
    const unsigned short* __restrict__ A, int lda,
    const unsigned short* __restrict__ W,
    int K,
    const float* __restrict__ biasA,
    const float* __restrict__ biasB,
    const float* __restrict__ Cinit,
    float* __restrict__ C)
{
    __shared__ __align__(16) unsigned short sA[2][128 * LDSA];
    __shared__ __align__(16) unsigned short sB[2][64 * LDSA];

    const int tid  = threadIdx.x;
    const int wv   = tid >> 5;
    const int lane = tid & 31;
    const int lh   = lane >> 4;      // half (0: lanes 0-15, 1: lanes 16-31)
    const int lr   = lane & 15;

    const int bm = blockIdx.y * 128;
    const int bn = blockIdx.x * 64;
    const int wm = (wv & 3) * 32;    // wave row offset inside block tile
    const int wn = (wv >> 2) * 32;   // wave col offset inside block tile

    // A staging: 512 chunks of 8 bf16 (16B). Thread handles chunk tid (rows
    // 0..63) and tid+256 (rows 64..127).
    const int caRow = tid >> 2;          // 0..63
    const int caOff = (tid & 3) * 8;     // 0,8,16,24
    // W staging: 256 chunks of 8 bf16; chunk t: k = t>>3, n0 = (t&7)*8.
    const int cbK = tid >> 3;            // 0..31
    const int cbN = (tid & 7) * 8;       // 0..56

    v8f acc00 = {0,0,0,0,0,0,0,0};
    v8f acc01 = {0,0,0,0,0,0,0,0};
    v8f acc10 = {0,0,0,0,0,0,0,0};
    v8f acc11 = {0,0,0,0,0,0,0,0};

    const int nk = K >> 5;

    auto load_stage = [&](int kt, int buf) {
        const unsigned short* ga0 = A + (size_t)(bm + caRow)      * (size_t)lda + (size_t)kt * 32 + caOff;
        const unsigned short* ga1 = A + (size_t)(bm + caRow + 64) * (size_t)lda + (size_t)kt * 32 + caOff;
        unsigned short* la0 = &sA[buf][caRow * LDSA + caOff];
        unsigned short* la1 = &sA[buf][(caRow + 64) * LDSA + caOff];
#if USE_ASYNC
        __builtin_amdgcn_global_load_async_to_lds_b128(
            (__attribute__((address_space(1))) v4i*)(unsigned long long)ga0,
            (__attribute__((address_space(3))) v4i*)(unsigned long long)la0, 0, 0);
        __builtin_amdgcn_global_load_async_to_lds_b128(
            (__attribute__((address_space(1))) v4i*)(unsigned long long)ga1,
            (__attribute__((address_space(3))) v4i*)(unsigned long long)la1, 0, 0);
#else
        *(u32x4*)la0 = *(const u32x4*)ga0;
        *(u32x4*)la1 = *(const u32x4*)ga1;
#endif
        // W tile [32 x 64] -> transposed into LDS sB[col][k]
        u32x4 wvv = *(const u32x4*)(W + (size_t)(kt * 32 + cbK) * 1024u + (size_t)(bn + cbN));
#pragma unroll
        for (int j = 0; j < 4; ++j) {
            unsigned int d = wvv[j];
            sB[buf][(cbN + 2 * j)     * LDSA + cbK] = (unsigned short)(d & 0xFFFFu);
            sB[buf][(cbN + 2 * j + 1) * LDSA + cbK] = (unsigned short)(d >> 16);
        }
    };

    load_stage(0, 0);

    for (int kt = 0; kt < nk; ++kt) {
        const int buf = kt & 1;
#if USE_ASYNC
        __builtin_amdgcn_s_wait_asynccnt(0);
#endif
        __syncthreads();
        if (kt + 1 < nk) load_stage(kt + 1, buf ^ 1);

        // --- A fragments (ISA 16-bit A 16x32 layout) ---
        // lane (lh,lr): row = lr, K = [lh*8 .. lh*8+7] and [16+lh*8 .. 16+lh*8+7]
        Frag a0, a1, b0, b1;
        const unsigned short* ap = &sA[buf][(wm + lr) * LDSA + lh * 8];
        a0.lo = *(const u32x4*)(ap);
        a0.hi = *(const u32x4*)(ap + 16);
        a1.lo = *(const u32x4*)(ap + 16 * LDSA);
        a1.hi = *(const u32x4*)(ap + 16 * LDSA + 16);
        // --- B fragments (same K order as A: K = [lh*8 .. lh*8+7] and [16+lh*8 .. 16+lh*8+7]) ---
        const unsigned short* bp = &sB[buf][(wn + lr) * LDSA + lh * 8];
        b0.lo = *(const u32x4*)(bp);
        b0.hi = *(const u32x4*)(bp + 16);
        b1.lo = *(const u32x4*)(bp + 16 * LDSA);
        b1.hi = *(const u32x4*)(bp + 16 * LDSA + 16);

        acc00 = __builtin_amdgcn_wmma_f32_16x16x32_bf16(false, frag_cast(a0), false, frag_cast(b0), (short)0, acc00, false, false);
        acc01 = __builtin_amdgcn_wmma_f32_16x16x32_bf16(false, frag_cast(a0), false, frag_cast(b1), (short)0, acc01, false, false);
        acc10 = __builtin_amdgcn_wmma_f32_16x16x32_bf16(false, frag_cast(a1), false, frag_cast(b0), (short)0, acc10, false, false);
        acc11 = __builtin_amdgcn_wmma_f32_16x16x32_bf16(false, frag_cast(a1), false, frag_cast(b1), (short)0, acc11, false, false);
    }

    // --- Epilogue: C/D layout -> lane(lh,lr), VGPR v: M = v + 8*lh, N = lr ---
    auto store_tile = [&](const v8f& acc, int tm, int tn) {
        const int col = bn + tn + lr;
        float bsum = 0.0f;
        if (biasA) bsum += biasA[col];
        if (biasB) bsum += biasB[col];
        const int row0 = bm + tm + lh * 8;
#pragma unroll
        for (int v = 0; v < 8; ++v) {
            size_t idx = (size_t)(row0 + v) * 1024u + (size_t)col;
            float val = acc[v] + bsum;
            if (Cinit) val += Cinit[idx];
            C[idx] = val;
        }
    };
    store_tile(acc00, wm,      wn);
    store_tile(acc01, wm,      wn + 16);
    store_tile(acc10, wm + 16, wn);
    store_tile(acc11, wm + 16, wn + 16);
}

// ---------------------------------------------------------------------------
// Gate fusion 1: forget/input/cell/e/r.  Writes c_new,e_new,r_new to d_out and
// bf16 [e_new | r_new] (4096 x 2048) for the phase-2 GEMM.
// ---------------------------------------------------------------------------
__global__ void gates1_kernel(const float* __restrict__ G0, const float* __restrict__ G1,
                              const float* __restrict__ G2, const float* __restrict__ EP,
                              const float* __restrict__ RP, const float* __restrict__ c_in,
                              float* __restrict__ out_c, float* __restrict__ out_e,
                              float* __restrict__ out_r, unsigned short* __restrict__ enrn) {
    unsigned int i = blockIdx.x * 256u + threadIdx.x;     // exact 4096*1024
    float fg = sigmoid_(G0[i]);
    float ig = sigmoid_(G1[i]);
    float cn = fg * c_in[i] + ig * softsign_(G2[i]);
    float en = softsign_(EP[i]);
    float rn = softsign_(RP[i]);
    out_c[i] = cn;
    out_e[i] = en;
    out_r[i] = rn;
    unsigned int row = i >> 10, col = i & 1023u;
    enrn[(size_t)row * 2048u + col]          = f2bf(en);
    enrn[(size_t)row * 2048u + 1024u + col]  = f2bf(rn);
}

// ---------------------------------------------------------------------------
// Gate fusion 2: output gate + h_new.
// ---------------------------------------------------------------------------
__global__ void gates2_kernel(const float* __restrict__ OP, const float* __restrict__ c_new,
                              float* __restrict__ out_h) {
    unsigned int i = blockIdx.x * 256u + threadIdx.x;     // exact 4096*1024
    float og = sigmoid_(OP[i]);
    out_h[i] = og * softsign_(c_new[i]);
}

// ---------------------------------------------------------------------------
// Host launcher
// ---------------------------------------------------------------------------
extern "C" void kernel_launch(void* const* d_in, const int* in_sizes, int n_in,
                              void* d_out, int out_size, void* d_ws, size_t ws_size,
                              hipStream_t stream) {
    (void)in_sizes; (void)n_in; (void)out_size; (void)ws_size;

    const float* x    = (const float*)d_in[0];
    const float* h    = (const float*)d_in[1];
    const float* c    = (const float*)d_in[2];
    const float* e    = (const float*)d_in[3];
    const float* r    = (const float*)d_in[4];
    const float* w_ii = (const float*)d_in[5];   // [512, 6144]
    const float* w_hh = (const float*)d_in[6];   // [1024, 6144]
    const float* w_cc = (const float*)d_in[7];   // [1024, 3072]
    const float* w_ee = (const float*)d_in[8];   // [1024, 4096]
    const float* w_rr = (const float*)d_in[9];   // [1024, 4096]
    const float* b_ih = (const float*)d_in[10];  // [6144]
    const float* b_hh = (const float*)d_in[11];  // [6144]

    const size_t B = 4096, H = 1024, CH = B * H;
    float* out   = (float*)d_out;
    float* out_h = out;
    float* out_c = out + CH;
    float* out_e = out + 2 * CH;
    float* out_r = out + 3 * CH;

    // ---- workspace layout ----
    char* ws = (char*)d_ws;
    size_t off = 0;
    auto take = [&](size_t bytes) -> void* {
        void* p = ws + off;
        off += (bytes + 255u) & ~(size_t)255u;
        return p;
    };
    unsigned short* Act = (unsigned short*)take(B * 4608 * 2);       // [4096 x 4608]
    unsigned short* P0  = (unsigned short*)take(4608 * 1024 * 2);    // forget-gate weights
    unsigned short* P1  = (unsigned short*)take(4608 * 1024 * 2);    // input-gate weights
    unsigned short* P2  = (unsigned short*)take(1536 * 1024 * 2);    // wb2 (x|h)
    unsigned short* P3  = (unsigned short*)take(1536 * 1024 * 2);    // wb3 (x|h)
    unsigned short* P4  = (unsigned short*)take(1536 * 1024 * 2);    // wb4 (x|h)
    unsigned short* WE2 = (unsigned short*)take(1024 * 1024 * 2);    // w_ee[:,2H:3H]
    unsigned short* WR2 = (unsigned short*)take(1024 * 1024 * 2);    // w_rr[:,2H:3H]
    unsigned short* P5  = (unsigned short*)take(2048 * 1024 * 2);    // [wc2+we3 ; wr3]
    float* G0 = (float*)take(CH * 4);
    float* G1 = (float*)take(CH * 4);
    float* G2 = (float*)take(CH * 4);
    float* G3 = (float*)take(CH * 4);
    float* G4 = (float*)take(CH * 4);
    float* EP = (float*)take(CH * 4);
    float* RP = (float*)take(CH * 4);
    unsigned short* ENRN = (unsigned short*)take(B * 2048 * 2);      // [4096 x 2048]
    float* OP = G0;   // phase-2 pre-activation reuses G0 (dead by then)

    // ---- 1. activation conversion ----
    convert_act_kernel<<<(B * 4608) / 256, 256, 0, stream>>>(x, h, c, e, r, Act);

    // ---- 2. weight packing ----
    auto pack = [&](const float* s, int ld, int c0, int rows, unsigned short* dst) {
        pack_w_kernel<<<(rows * 1024) / 256, 256, 0, stream>>>(s, ld, c0, nullptr, 0, 0, dst);
    };
    for (int g = 0; g < 2; ++g) {
        unsigned short* Pg = (g == 0) ? P0 : P1;
        pack(w_ii, 6144, g * 1024,  512, Pg);
        pack(w_hh, 6144, g * 1024, 1024, Pg +  512 * 1024);
        pack(w_cc, 3072, g * 1024, 1024, Pg + 1536 * 1024);
        pack(w_ee, 4096, g * 1024, 1024, Pg + 2560 * 1024);
        pack(w_rr, 4096, g * 1024, 1024, Pg + 3584 * 1024);
    }
    unsigned short* Pxh[3] = {P2, P3, P4};
    for (int g = 2; g <= 4; ++g) {
        pack(w_ii, 6144, g * 1024,  512, Pxh[g - 2]);
        pack(w_hh, 6144, g * 1024, 1024, Pxh[g - 2] + 512 * 1024);
    }
    pack(w_ee, 4096, 2048, 1024, WE2);
    pack(w_rr, 4096, 2048, 1024, WR2);
    // P5 rows 0:1024 = wc2 + we3 (both multiply e_new); rows 1024:2048 = wr3
    pack_w_kernel<<<(1024 * 1024) / 256, 256, 0, stream>>>(w_cc, 3072, 2048, w_ee, 4096, 3072, P5);
    pack(w_rr, 4096, 3072, 1024, P5 + 1024 * 1024);

    // ---- 3. phase-1 GEMMs ----
    dim3 ggrid(1024 / 64, 4096 / 128);   // (N blocks, M blocks)
    auto gemm = [&](const unsigned short* Ap, int lda, const unsigned short* Wp, int K,
                    const float* ba, const float* bb, const float* ci, float* Cp) {
        gemm_bf16_kernel<<<ggrid, 256, 0, stream>>>(Ap, lda, Wp, K, ba, bb, ci, Cp);
    };
    gemm(Act,        4608, P0, 4608, b_ih + 0 * H, b_hh + 0 * H, nullptr, G0);  // forget pre-act
    gemm(Act,        4608, P1, 4608, b_ih + 1 * H, b_hh + 1 * H, nullptr, G1);  // input pre-act
    gemm(Act,        4608, P2, 1536, b_ih + 2 * H, b_hh + 2 * H, nullptr, G2);  // wb2
    gemm(Act,        4608, P3, 1536, b_ih + 3 * H, b_hh + 3 * H, nullptr, G3);  // wb3
    gemm(Act,        4608, P4, 1536, b_ih + 4 * H, b_hh + 4 * H, nullptr, G4);  // wb4
    gemm(Act + 2560, 4608, WE2, 1024, nullptr, nullptr, G3, EP);                // wb3 + e@we2
    gemm(Act + 3584, 4608, WR2, 1024, nullptr, nullptr, G3, RP);                // wb3 + r@wr2

    // ---- 4. gate fusion 1: c_new, e_new, r_new (+ bf16 repack) ----
    gates1_kernel<<<CH / 256, 256, 0, stream>>>(G0, G1, G2, EP, RP, c, out_c, out_e, out_r, ENRN);

    // ---- 5. phase-2 GEMM: wb4 + [e_new|r_new] @ [wc2+we3 ; wr3] ----
    gemm(ENRN, 2048, P5, 2048, nullptr, nullptr, G4, OP);

    // ---- 6. gate fusion 2: h_new ----
    gates2_kernel<<<CH / 256, 256, 0, stream>>>(OP, out_c, out_h);
}
